// MultiHeadedAttention_63823214019124
// MI455X (gfx1250) — matrix operand order repaired and checked by reading, not compiled, not run on hardware
//
#include <hip/hip_runtime.h>
#include <hip/hip_bf16.h>
#include <cstdint>
#include <cstddef>

// ---------------------------------------------------------------------------
// MI455X (gfx1250) GQA attention block.
// - all matmuls: v_wmma_f32_16x16x32_f16
// - GEMM LDS staging: TENSOR_LOAD_TO_LDS (TDM, TENSORcnt) when available,
//   else GLOBAL_LOAD_ASYNC_TO_LDS_B128 (ASYNCcnt); double-buffered k-loop
// - flash-attn staging: GLOBAL_LOAD_ASYNC_TO_LDS_B128
// ~155 GFLOP vs ~110MB HBM traffic -> compute-bound -> matrix engine + DMA.
// ---------------------------------------------------------------------------

#define BATCH 2
#define SEQ   2048
#define HIDD  2048
#define NHQ   16
#define NKVH  4
#define DHEAD 128

typedef _Float16 v8h  __attribute__((ext_vector_type(8)));
typedef _Float16 v16h __attribute__((ext_vector_type(16)));
typedef float    v8f  __attribute__((ext_vector_type(8)));

__device__ __forceinline__ v8f zero_v8f() {
    v8f z;
#pragma unroll
    for (int i = 0; i < 8; ++i) z[i] = 0.0f;
    return z;
}

__device__ __forceinline__ v16h combine16(v8h lo, v8h hi) {
    return __builtin_shufflevector(lo, hi, 0,1,2,3,4,5,6,7,8,9,10,11,12,13,14,15);
}

// D = A(16x32 f16) * B(32x16 f16) + C(16x16 f32)
__device__ __forceinline__ v8f wmma16(v16h a, v16h b, v8f c) {
    return __builtin_amdgcn_wmma_f32_16x16x32_f16(false, a, false, b,
                                                  (short)0, c, false, false);
}

// 16B global -> LDS direct copy, no VGPR round-trip, tracked by ASYNCcnt.
__device__ __forceinline__ void async_cp16(const void* gsrc, const void* ldst) {
    asm volatile("global_load_async_to_lds_b128 %0, %1, off"
                 :
                 : "v"((uint32_t)(uintptr_t)ldst),
                   "v"((uint64_t)(uintptr_t)gsrc)
                 : "memory");
}

__device__ __forceinline__ void wait_async0() {
#if __has_builtin(__builtin_amdgcn_s_wait_asynccnt)
    __builtin_amdgcn_s_wait_asynccnt(0);
#else
    asm volatile("s_wait_asynccnt 0x0" ::: "memory");
#endif
}

// ---------------------------------------------------------------------------
// Tensor Data Mover: one-instruction 2D tile DMA global->LDS with hardware
// LDS row padding (ISA ch.8 D# descriptor). Wave-level op, TENSORcnt-tracked.
// This toolchain exposes the 6-arg builtin:
//   (uint32x4 g0, int32x8 g1, int32x4 g2, int32x4 g3, int32x8 pad, i32 cpol)
// ---------------------------------------------------------------------------
#if __has_builtin(__builtin_amdgcn_tensor_load_to_lds) && \
    __has_builtin(__builtin_amdgcn_s_wait_tensorcnt)
#define HAS_TDM 1
typedef unsigned int u32x4 __attribute__((ext_vector_type(4)));
typedef int          i32x4 __attribute__((ext_vector_type(4)));
typedef int          i32x8 __attribute__((ext_vector_type(8)));

// tile_w/tile_h in 2-byte elements; stride = row stride in elements;
// pad_int_enc: LDS pad interval (0=2,1=4,2=8,3=16,4=32,5=64,... DWORDs);
// pad_amt_enc: pad amount minus 1 in DWORDs.
__device__ __forceinline__ void tdm_load_2d(const void* ldst, const void* gsrc,
                                            uint32_t tile_w, uint32_t tile_h,
                                            uint32_t stride,
                                            uint32_t pad_int_enc,
                                            uint32_t pad_amt_enc) {
    const uint64_t ga = (uint64_t)(uintptr_t)gsrc;
    u32x4 g0;
    g0[0] = 1u;                                            // count=1 (valid D#)
    g0[1] = (uint32_t)(uintptr_t)ldst;                     // lds_addr
    g0[2] = (uint32_t)ga;                                  // global_addr[31:0]
    g0[3] = (uint32_t)((ga >> 32) & 0x01FFFFFFu) | (2u << 30);  // ga[56:32]|type=2
    i32x8 g1;
    g1[0] = (int)((1u << 16)                               // data_size = 2 bytes
                | (1u << 20)                               // pad_enable
                | (pad_int_enc << 22) | (pad_amt_enc << 25));
    g1[1] = (int)(tile_w << 16);                           // tensor_dim0[15:0]
    g1[2] = (int)(tile_h << 16);                           // td0 hi=0 | tensor_dim1 lo
    g1[3] = (int)(tile_w << 16);                           // td1 hi=0 | tile_dim0
    g1[4] = (int)(tile_h & 0xFFFFu);                       // tile_dim1 | tile_dim2=0
    g1[5] = (int)stride;                                   // tensor_dim0_stride lo
    g1[6] = 0;                                             // stride hi | dim1_stride lo
    g1[7] = 0;
    i32x4 z4; z4[0] = z4[1] = z4[2] = z4[3] = 0;           // 2D: groups 2/3 unused
    i32x8 z8;
#pragma unroll
    for (int i = 0; i < 8; ++i) z8[i] = 0;
    __builtin_amdgcn_tensor_load_to_lds(g0, g1, z4, z4, z8, 0);
}
#else
#define HAS_TDM 0
#endif

// ---------------------------------------------------------------------------
// fp32 -> f16 elementwise convert
// ---------------------------------------------------------------------------
__global__ __launch_bounds__(256) void cvt_f32_f16_kernel(
        const float* __restrict__ src, _Float16* __restrict__ dst, int n) {
    int i = blockIdx.x * 256 + threadIdx.x;
    if (i < n) dst[i] = (_Float16)src[i];
}

// ---------------------------------------------------------------------------
// fp32 KxN -> f16 NxK transpose-convert
// ---------------------------------------------------------------------------
__global__ __launch_bounds__(256) void transpose_cvt_kernel(
        const float* __restrict__ src, _Float16* __restrict__ dst, int K, int N) {
    __shared__ float tile[32][33];
    const int bx = blockIdx.x * 32;           // n
    const int by = blockIdx.y * 32;           // k
    const int tx = threadIdx.x & 31;
    const int ty = threadIdx.x >> 5;          // 0..7
#pragma unroll
    for (int i = 0; i < 32; i += 8)
        tile[ty + i][tx] = src[(size_t)(by + ty + i) * N + bx + tx];
    __syncthreads();
#pragma unroll
    for (int i = 0; i < 32; i += 8)
        dst[(size_t)(bx + ty + i) * K + by + tx] = (_Float16)tile[tx][ty + i];
}

// ---------------------------------------------------------------------------
// Double-buffered WMMA GEMM: C(MxN) = A(MxK f16 row-major) * Bt(NxK f16)^T
// 256 threads = 8 waves; tile 128x128; wave = 64x32 (4x2 fragments).
// Pipeline: TDM (or async) fills buf^1 while WMMAs consume buf.
// mode 0: f16 out; mode 1: f32 out; mode 2: f16 V-transposed out.
// ---------------------------------------------------------------------------
__global__ __launch_bounds__(256) void gemm_f16_wmma_kernel(
        const _Float16* __restrict__ A, const _Float16* __restrict__ Bt,
        void* __restrict__ Cout, int M, int N, int K, int mode, int Sdim) {
    // 40-half row stride = 80B = 5x16B: 16B-aligned b128 reads; for TDM the
    // padding is reproduced with pad_interval=16 DWORDs, pad_amount=4 DWORDs.
    __shared__ __align__(16) _Float16 tA[2][128][40];
    __shared__ __align__(16) _Float16 tB[2][128][40];

    const int tid  = threadIdx.x;
    const int wave = tid >> 5;
    const int lane = tid & 31;
    const int l15  = lane & 15;
    const int lhi  = lane >> 4;
    const int wm   = wave >> 2;   // 0..1  -> 64-row slab
    const int wn   = wave & 3;    // 0..3  -> 32-col slab
    const int mb   = blockIdx.x * 128;
    const int nb   = blockIdx.y * 128;

#if HAS_TDM
    // scalar-branch issuer selection (TDM ignores EXEC; must branch, not mask)
    const bool issuer = (__builtin_amdgcn_readfirstlane(wave) == 0);
#else
    const int c0   = tid;               // chunk ids 0..511 (16B chunks)
    const int row0 = c0 >> 2,       off0 = (c0 & 3) * 8;
    const int c1   = tid + 256;
    const int row1 = c1 >> 2,       off1 = (c1 & 3) * 8;
#endif

    v8f acc[4][2];
#pragma unroll
    for (int mi = 0; mi < 4; ++mi)
#pragma unroll
        for (int ni = 0; ni < 2; ++ni) acc[mi][ni] = zero_v8f();

    // prologue: fill buffer 0 with the first K-slab
#if HAS_TDM
    if (issuer) {
        tdm_load_2d(&tA[0][0][0], &A[(size_t)mb * K], 32, 128, (uint32_t)K, 3, 3);
        tdm_load_2d(&tB[0][0][0], &Bt[(size_t)nb * K], 32, 128, (uint32_t)K, 3, 3);
        __builtin_amdgcn_s_wait_tensorcnt(0);
    }
    __syncthreads();
#else
    async_cp16(&A[(size_t)(mb + row0) * K + off0], &tA[0][row0][off0]);
    async_cp16(&Bt[(size_t)(nb + row0) * K + off0], &tB[0][row0][off0]);
    async_cp16(&A[(size_t)(mb + row1) * K + off1], &tA[0][row1][off1]);
    async_cp16(&Bt[(size_t)(nb + row1) * K + off1], &tB[0][row1][off1]);
    wait_async0();
    __syncthreads();
#endif

    for (int kb = 0; kb < K; kb += 32) {
        const int cur = (kb >> 5) & 1;
        // kick off next K-slab into the other buffer (overlaps the WMMAs)
        if (kb + 32 < K) {
            const int nxt = cur ^ 1;
            const int kn  = kb + 32;
#if HAS_TDM
            if (issuer) {
                tdm_load_2d(&tA[nxt][0][0], &A[(size_t)mb * K + kn], 32, 128,
                            (uint32_t)K, 3, 3);
                tdm_load_2d(&tB[nxt][0][0], &Bt[(size_t)nb * K + kn], 32, 128,
                            (uint32_t)K, 3, 3);
            }
#else
            async_cp16(&A[(size_t)(mb + row0) * K + kn + off0], &tA[nxt][row0][off0]);
            async_cp16(&Bt[(size_t)(nb + row0) * K + kn + off0], &tB[nxt][row0][off0]);
            async_cp16(&A[(size_t)(mb + row1) * K + kn + off1], &tA[nxt][row1][off1]);
            async_cp16(&Bt[(size_t)(nb + row1) * K + kn + off1], &tB[nxt][row1][off1]);
#endif
        }

#pragma unroll
        for (int mi = 0; mi < 4; ++mi) {
            // A fragment (ISA 7.12.2): lane m=l15, halves {8*kg+0..7, 16+8*kg+0..7}
            const int am = wm * 64 + mi * 16 + l15;
            v16h af = combine16(*(const v8h*)&tA[cur][am][8 * lhi],
                                *(const v8h*)&tA[cur][am][16 + 8 * lhi]);
#pragma unroll
            for (int ni = 0; ni < 2; ++ni) {
                // B fragment: lane n=l15, halves {8*kg+0..7, 16+8*kg+0..7}
                const int bn = wn * 32 + ni * 16 + l15;
                v16h bf = combine16(*(const v8h*)&tB[cur][bn][8 * lhi],
                                    *(const v8h*)&tB[cur][bn][16 + 8 * lhi]);
                acc[mi][ni] = wmma16(af, bf, acc[mi][ni]);
            }
        }
#if HAS_TDM
        if (issuer) __builtin_amdgcn_s_wait_tensorcnt(0);
#else
        wait_async0();       // next tile fully landed in LDS
#endif
        __syncthreads();     // everyone done reading cur; buffers rotate
    }

    // epilogue: C/D layout -> element (M = r + 8*lhi, N = l15)
#pragma unroll
    for (int mi = 0; mi < 4; ++mi) {
#pragma unroll
        for (int ni = 0; ni < 2; ++ni) {
            const int col = nb + wn * 32 + ni * 16 + l15;
#pragma unroll
            for (int r = 0; r < 8; ++r) {
                const int Mrow = mb + wm * 64 + mi * 16 + r + 8 * lhi;
                const float v = acc[mi][ni][r];
                if (mode == 0) {
                    ((_Float16*)Cout)[(size_t)Mrow * N + col] = (_Float16)v;
                } else if (mode == 1) {
                    ((float*)Cout)[(size_t)Mrow * N + col] = v;
                } else {
                    const int b = Mrow / Sdim, s = Mrow % Sdim;
                    ((_Float16*)Cout)[((size_t)b * N + col) * Sdim + s] = (_Float16)v;
                }
            }
        }
    }
}

// ---------------------------------------------------------------------------
// Interleaved RoPE, in place on f16 [rows][heads*128]
// ---------------------------------------------------------------------------
__global__ __launch_bounds__(256) void rope_kernel(
        _Float16* __restrict__ t, const float* __restrict__ cosb,
        const float* __restrict__ sinb, int rows, int heads) {
    int idx = blockIdx.x * 256 + threadIdx.x;
    int total = rows * heads * (DHEAD / 2);
    if (idx >= total) return;
    int f   = idx & 63;
    int tmp = idx >> 6;
    int hh  = tmp % heads;
    int row = tmp / heads;
    int pos = row & (SEQ - 1);
    float c = cosb[pos * 64 + f], s = sinb[pos * 64 + f];
    size_t base = (size_t)row * heads * DHEAD + hh * DHEAD + 2 * f;
    float te = (float)t[base], to = (float)t[base + 1];
    t[base]     = (_Float16)(te * c - to * s);
    t[base + 1] = (_Float16)(to * c + te * s);
}

// ---------------------------------------------------------------------------
// Flash attention (causal, GQA rep=4). 256 thr = 8 waves, q-tile 128; each
// wave owns 16 q rows (wave-private softmax stats). K/Vt tiles staged via
// GLOBAL_LOAD_ASYNC_TO_LDS_B128; scores & P*V via WMMA.
// ---------------------------------------------------------------------------
__global__ __launch_bounds__(256) void flash_attn_kernel(
        const _Float16* __restrict__ Q, const _Float16* __restrict__ Km,
        const _Float16* __restrict__ Vt, _Float16* __restrict__ CTX) {
    __shared__ __align__(16) _Float16 tK[64][136];    // [kv row][d]      (+8 pad)
    __shared__ __align__(16) _Float16 tV[128][72];    // [d][kv col]      (+8 pad)
    __shared__ __align__(16) _Float16 tP[8][16][72];  // per-wave P tile  (+8 pad)

    const int tid  = threadIdx.x;
    const int wave = tid >> 5;
    const int lane = tid & 31;
    const int l15  = lane & 15;
    const int lhi  = lane >> 4;
    const int qb   = blockIdx.x;          // S/128 tiles
    const int h    = blockIdx.y;          // NH
    const int b    = blockIdx.z;          // BATCH
    const int kvh  = h >> 2;              // NH/NKV = 4
    const int qg   = qb * 128 + wave * 16;        // this wave's first q row
    const size_t row0 = (size_t)b * SEQ + qg;     // global row of wave's q block

    // Preload Q A-fragments for the whole D=128 (4 x K=32 windows)
    v16h qf[4];
    {
        const _Float16* qrow = Q + ((size_t)row0 + l15) * (NHQ * DHEAD) + h * DHEAD;
#pragma unroll
        for (int kk = 0; kk < 4; ++kk)
            qf[kk] = combine16(*(const v8h*)&qrow[32 * kk + 8 * lhi],
                               *(const v8h*)&qrow[32 * kk + 16 + 8 * lhi]);
    }

    v8f ctx[8];
#pragma unroll
    for (int nd = 0; nd < 8; ++nd) ctx[nd] = zero_v8f();
    float mrun[8], lrun[8];
#pragma unroll
    for (int r = 0; r < 8; ++r) { mrun[r] = -3.0e38f; lrun[r] = 0.0f; }

    const float scale = 0.08838834764831845f;     // 1/sqrt(128)
    const int nkt = qb * 2 + 2;                   // causal: cover k <= qb*128+127

    for (int kt = 0; kt < nkt; ++kt) {
        const int kbase = kt * 64;
        // async-stage K tile (64 x 128 halves) and Vt tile (128 x 64 halves)
#pragma unroll
        for (int i = 0; i < 4; ++i) {
            int c   = tid + i * 256;              // 0..1023
            int row = c >> 4;                     // 0..63
            int off = (c & 15) * 8;               // 0..120
            async_cp16(&Km[((size_t)b * SEQ + kbase + row) * (NKVH * DHEAD)
                           + kvh * DHEAD + off],
                       &tK[row][off]);
        }
#pragma unroll
        for (int i = 0; i < 4; ++i) {
            int c   = tid + i * 256;
            int row = c >> 3;                     // 0..127 (d)
            int off = (c & 7) * 8;                // 0..56
            async_cp16(&Vt[((size_t)b * (NKVH * DHEAD) + kvh * DHEAD + row) * SEQ
                           + kbase + off],
                       &tV[row][off]);
        }
        wait_async0();
        __syncthreads();

        if (kbase <= qg + 15) {   // wave-uniform causal skip
            // ---- scores: 16x64 = 4 fragments, K-dim = 128 (4 WMMAs each) ----
            v8f sf[4];
#pragma unroll
            for (int ni = 0; ni < 4; ++ni) {
                sf[ni] = zero_v8f();
#pragma unroll
                for (int kk = 0; kk < 4; ++kk) {
                    v16h bf = combine16(
                        *(const v8h*)&tK[ni * 16 + l15][32 * kk + 8 * lhi],
                        *(const v8h*)&tK[ni * 16 + l15][32 * kk + 16 + 8 * lhi]);
                    sf[ni] = wmma16(qf[kk], bf, sf[ni]);
                }
            }
            // ---- mask + online softmax (rows live in 16-lane groups) ----
#pragma unroll
            for (int r = 0; r < 8; ++r) {
                const int qrow = qg + r + 8 * lhi;
                float rowmax = -3.0e38f;
#pragma unroll
                for (int ni = 0; ni < 4; ++ni) {
                    const int col = kbase + ni * 16 + l15;
                    float v = sf[ni][r] * scale;
                    v = (col <= qrow) ? v : -3.0e38f;
                    sf[ni][r] = v;
                    rowmax = fmaxf(rowmax, v);
                }
#pragma unroll
                for (int o = 8; o > 0; o >>= 1)
                    rowmax = fmaxf(rowmax, __shfl_xor(rowmax, o, 16));
                const float mnew = fmaxf(mrun[r], rowmax);
                const float corr = __expf(mrun[r] - mnew);
                float rs = 0.0f;
#pragma unroll
                for (int ni = 0; ni < 4; ++ni) {
                    float p = __expf(sf[ni][r] - mnew);
                    sf[ni][r] = p;
                    rs += p;
                }
#pragma unroll
                for (int o = 8; o > 0; o >>= 1) rs += __shfl_xor(rs, o, 16);
                lrun[r] = lrun[r] * corr + rs;
                mrun[r] = mnew;
#pragma unroll
                for (int nd = 0; nd < 8; ++nd) ctx[nd][r] *= corr;
            }
            // ---- re-shape P: C-fragment -> LDS -> A-fragment (wave-private) ----
#pragma unroll
            for (int ni = 0; ni < 4; ++ni)
#pragma unroll
                for (int r = 0; r < 8; ++r)
                    tP[wave][r + 8 * lhi][ni * 16 + l15] = (_Float16)sf[ni][r];
            // ---- ctx += P(16x64) @ V(64x128) ----
#pragma unroll
            for (int kk2 = 0; kk2 < 2; ++kk2) {
                v16h pf = combine16(
                    *(const v8h*)&tP[wave][l15][32 * kk2 + 8 * lhi],
                    *(const v8h*)&tP[wave][l15][32 * kk2 + 16 + 8 * lhi]);
#pragma unroll
                for (int nd = 0; nd < 8; ++nd) {
                    v16h vf = combine16(
                        *(const v8h*)&tV[nd * 16 + l15][32 * kk2 + 8 * lhi],
                        *(const v8h*)&tV[nd * 16 + l15][32 * kk2 + 16 + 8 * lhi]);
                    ctx[nd] = wmma16(pf, vf, ctx[nd]);
                }
            }
        }
        __syncthreads();
    }

    // epilogue: normalize and store f16 context [B*S][NH*128]
#pragma unroll
    for (int r = 0; r < 8; ++r) lrun[r] = 1.0f / lrun[r];
#pragma unroll
    for (int nd = 0; nd < 8; ++nd) {
        const int col = h * DHEAD + nd * 16 + l15;
#pragma unroll
        for (int r = 0; r < 8; ++r) {
            const size_t row = row0 + r + 8 * lhi;
            CTX[row * (NHQ * DHEAD) + col] = (_Float16)(ctx[nd][r] * lrun[r]);
        }
    }
}

// ---------------------------------------------------------------------------
// Host launcher
// ---------------------------------------------------------------------------
extern "C" void kernel_launch(void* const* d_in, const int* in_sizes, int n_in,
                              void* d_out, int out_size, void* d_ws, size_t ws_size,
                              hipStream_t stream) {
    const float* x   = (const float*)d_in[0];
    const float* wq  = (const float*)d_in[1];
    const float* wk  = (const float*)d_in[2];
    const float* wv  = (const float*)d_in[3];
    const float* wo  = (const float*)d_in[4];
    const float* cosb = (const float*)d_in[5];
    const float* sinb = (const float*)d_in[6];
    // d_in[7] = mask (causal tril, applied analytically in-kernel)
    float* out = (float*)d_out;

    const int ROWS = BATCH * SEQ;                  // 4096
    // workspace layout (f16 halves)
    _Float16* W = (_Float16*)d_ws;
    const size_t O_X16 = 0;                                       // 4096*2048
    const size_t O_WQT = O_X16 + (size_t)ROWS * HIDD;             // 2048*2048
    const size_t O_WKT = O_WQT + (size_t)HIDD * (NHQ * DHEAD);    // 512*2048
    const size_t O_WVT = O_WKT + (size_t)HIDD * (NKVH * DHEAD);
    const size_t O_WOT = O_WVT + (size_t)HIDD * (NKVH * DHEAD);   // 2048*2048
    const size_t O_Q   = O_WOT + (size_t)(NHQ * DHEAD) * HIDD;
    const size_t O_K   = O_Q + (size_t)ROWS * (NHQ * DHEAD);
    const size_t O_VT  = O_K + (size_t)ROWS * (NKVH * DHEAD);
    const size_t O_CTX = O_VT + (size_t)ROWS * (NKVH * DHEAD);
    _Float16 *x16 = W + O_X16, *wqt = W + O_WQT, *wkt = W + O_WKT,
             *wvt = W + O_WVT, *wot = W + O_WOT, *Q16 = W + O_Q,
             *K16 = W + O_K,  *VT  = W + O_VT,  *CTX = W + O_CTX;

    // 1) convert activations + transpose-convert weights to f16
    {
        int n = ROWS * HIDD;
        cvt_f32_f16_kernel<<<(n + 255) / 256, 256, 0, stream>>>(x, x16, n);
    }
    transpose_cvt_kernel<<<dim3((NHQ * DHEAD) / 32, HIDD / 32), 256, 0, stream>>>(wq, wqt, HIDD, NHQ * DHEAD);
    transpose_cvt_kernel<<<dim3((NKVH * DHEAD) / 32, HIDD / 32), 256, 0, stream>>>(wk, wkt, HIDD, NKVH * DHEAD);
    transpose_cvt_kernel<<<dim3((NKVH * DHEAD) / 32, HIDD / 32), 256, 0, stream>>>(wv, wvt, HIDD, NKVH * DHEAD);
    transpose_cvt_kernel<<<dim3(HIDD / 32, (NHQ * DHEAD) / 32), 256, 0, stream>>>(wo, wot, NHQ * DHEAD, HIDD);

    // 2) QKV projections (WMMA); V written transposed for the P*V stage
    gemm_f16_wmma_kernel<<<dim3(ROWS / 128, (NHQ * DHEAD) / 128), 256, 0, stream>>>(
        x16, wqt, Q16, ROWS, NHQ * DHEAD, HIDD, /*mode=*/0, 0);
    gemm_f16_wmma_kernel<<<dim3(ROWS / 128, (NKVH * DHEAD) / 128), 256, 0, stream>>>(
        x16, wkt, K16, ROWS, NKVH * DHEAD, HIDD, /*mode=*/0, 0);
    gemm_f16_wmma_kernel<<<dim3(ROWS / 128, (NKVH * DHEAD) / 128), 256, 0, stream>>>(
        x16, wvt, VT, ROWS, NKVH * DHEAD, HIDD, /*mode=*/2, SEQ);

    // 3) RoPE on Q and K
    {
        int nq = ROWS * NHQ * (DHEAD / 2);
        rope_kernel<<<(nq + 255) / 256, 256, 0, stream>>>(Q16, cosb, sinb, ROWS, NHQ);
        int nk = ROWS * NKVH * (DHEAD / 2);
        rope_kernel<<<(nk + 255) / 256, 256, 0, stream>>>(K16, cosb, sinb, ROWS, NKVH);
    }

    // 4) causal flash attention (WMMA scores + WMMA context)
    flash_attn_kernel<<<dim3(SEQ / 128, NHQ, BATCH), 256, 0, stream>>>(Q16, K16, VT, CTX);

    // 5) output projection, f32 out
    gemm_f16_wmma_kernel<<<dim3(ROWS / 128, HIDD / 128), 256, 0, stream>>>(
        CTX, wot, out, ROWS, HIDD, NHQ * DHEAD, /*mode=*/1, 0);
}
